// FlashRoPEAttention_87668872445997
// MI455X (gfx1250) — compile-verified, matrix operand order checked by reading, not run on hardware
//
#include <hip/hip_runtime.h>

// ---------------------------------------------------------------------------
// Types for CDNA5 WMMA (gfx1250, wave32)
// ---------------------------------------------------------------------------
typedef __attribute__((ext_vector_type(16))) __bf16       v16bf;
typedef __attribute__((ext_vector_type(8)))  float        v8f;
typedef __attribute__((ext_vector_type(4)))  unsigned int v4u;
typedef unsigned short ushort_t;

#define DEV static __device__ __forceinline__

// Problem constants (match reference)
constexpr int B_   = 2;
constexpr int L_   = 2048;
constexpr int D_   = 2048;
constexpr int H_   = 16;
constexpr int HKV_ = 4;
constexpr int HD_  = 128;
constexpr long M_ROWS = (long)B_ * L_;      // 4096 GEMM rows
constexpr float QK_SCALE = 0.08838834764831845f; // 1/sqrt(128)

union FragU { v16bf v; v4u c[2]; };

// Load a 16x32 bf16 A/B fragment chunk pair. `row` points at the start of the
// 16-bit row owned by this lane; elemOff selects the K-window:
//   VGPRs 0-3 <- 8 bf16 at elemOff        (16 bytes)
//   VGPRs 4-7 <- 8 bf16 at elemOff + 16   (16 bytes)
DEV v16bf load_frag(const ushort_t* row, int elemOff) {
  FragU f;
  f.c[0] = *(const v4u*)(row + elemOff);
  f.c[1] = *(const v4u*)(row + elemOff + 16);
  return f.v;
}

DEV v8f wmma_bf16(v16bf a, v16bf b, v8f c) {
  // D = A(16x32 bf16) * B(32x16 bf16) + C(16x16 f32)
  return __builtin_amdgcn_wmma_f32_16x16x32_bf16(
      /*neg_a=*/false, a, /*neg_b=*/false, b,
      /*c_mod=*/(short)0, c, /*reuse_a=*/false, /*reuse_b=*/false);
}

DEV ushort_t f2bf(float f) {  // round-to-nearest-even fp32 -> bf16
  unsigned u = __float_as_uint(f);
  unsigned r = ((u >> 16) & 1u) + 0x7FFFu;
  return (ushort_t)((u + r) >> 16);
}

DEV v8f zero8() {
  v8f z;
#pragma unroll
  for (int e = 0; e < 8; ++e) z[e] = 0.0f;
  return z;
}

// ---------------------------------------------------------------------------
// 1) fp32 -> bf16 convert (grid-stride)
// ---------------------------------------------------------------------------
__global__ void k_f32_to_bf16(const float* __restrict__ src,
                              ushort_t* __restrict__ dst, long n) {
  long i = (long)blockIdx.x * blockDim.x + threadIdx.x;
  long stride = (long)gridDim.x * blockDim.x;
  for (; i < n; i += stride) dst[i] = f2bf(src[i]);
}

// ---------------------------------------------------------------------------
// 2) bf16 WMMA GEMM:  C[M,N] = A[M,K] @ W[N,K]^T + bias[N]   (fp32 out)
//    Each wave computes a 32(M) x 64(N) tile; 4 waves / block (one per SIMD).
//    __launch_bounds__(128,1): keep accumulators register-resident (no spills).
// ---------------------------------------------------------------------------
__global__ __launch_bounds__(128, 1)
void k_gemm_bf16(const ushort_t* __restrict__ A,
                 const ushort_t* __restrict__ W,
                 const float* __restrict__ bias,
                 float* __restrict__ C,
                 int M, int N, int K) {
  const int lane  = threadIdx.x & 31;
  const int nlane = lane & 15;
  const int khalf = lane >> 4;      // also the "half" for C layout
  const int waveId = blockIdx.x * (blockDim.x >> 5) + (threadIdx.x >> 5);
  const int tilesN = N >> 6;
  const int tn = waveId % tilesN;
  const int tm = waveId / tilesN;
  if (tm >= (M >> 5)) return;

  v8f acc[2][4];
#pragma unroll
  for (int u = 0; u < 2; ++u)
#pragma unroll
    for (int t = 0; t < 4; ++t) acc[u][t] = zero8();

  const ushort_t* arow[2];
  const ushort_t* brow[4];
#pragma unroll
  for (int u = 0; u < 2; ++u)
    arow[u] = A + (size_t)(tm * 32 + u * 16 + nlane) * K;
#pragma unroll
  for (int t = 0; t < 4; ++t)
    brow[t] = W + (size_t)(tn * 64 + t * 16 + nlane) * K;

  for (int kk = 0; kk < K; kk += 32) {
    const int eo = kk + khalf * 8;
    v16bf af[2], bf[4];
#pragma unroll
    for (int u = 0; u < 2; ++u) af[u] = load_frag(arow[u], eo);
#pragma unroll
    for (int t = 0; t < 4; ++t) bf[t] = load_frag(brow[t], eo);
#pragma unroll
    for (int u = 0; u < 2; ++u)
#pragma unroll
      for (int t = 0; t < 4; ++t) acc[u][t] = wmma_bf16(af[u], bf[t], acc[u][t]);
  }

  // Epilogue: C-layout -> row m = khalf*8 + i, col n = nlane (within subtile)
#pragma unroll
  for (int u = 0; u < 2; ++u) {
#pragma unroll
    for (int t = 0; t < 4; ++t) {
      const int n = tn * 64 + t * 16 + nlane;
      const float bv = bias[n];
#pragma unroll
      for (int i = 0; i < 8; ++i) {
        const int m = tm * 32 + u * 16 + khalf * 8 + i;
        C[(size_t)m * N + n] = acc[u][t][i] + bv;
      }
    }
  }
}

// ---------------------------------------------------------------------------
// 3) RoPE + pack to bf16 head-major layout dst[b][h][l][hd]
//    One thread handles one rotation pair (i, i+64).
// ---------------------------------------------------------------------------
__global__ void k_rope_pack(const float* __restrict__ proj, int ld,
                            const float* __restrict__ cosT,
                            const float* __restrict__ sinT,
                            ushort_t* __restrict__ dst, int nH, float scale,
                            long total) {
  long idx = (long)blockIdx.x * blockDim.x + threadIdx.x;
  if (idx >= total) return;
  const int i   = (int)(idx & 63);
  long rem      = idx >> 6;
  const int h   = (int)(rem % nH); rem /= nH;
  const int l   = (int)(rem % L_);
  const int b   = (int)(rem / L_);

  const size_t rowf = ((size_t)b * L_ + l);
  const int col0 = h * HD_ + i;
  const float v0 = proj[rowf * ld + col0];
  const float v1 = proj[rowf * ld + col0 + 64];
  const size_t cs = rowf * HD_;
  const float c0 = cosT[cs + i],      s0 = sinT[cs + i];
  const float c1 = cosT[cs + i + 64], s1 = sinT[cs + i + 64];
  const float o0 = v0 * c0 - v1 * s0;
  const float o1 = v1 * c1 + v0 * s1;
  const size_t dbase = (((size_t)b * nH + h) * L_ + l) * HD_;
  dst[dbase + i]      = f2bf(o0 * scale);
  dst[dbase + i + 64] = f2bf(o1 * scale);
}

// ---------------------------------------------------------------------------
// 4) V: kvproj fp32 (b,l, 512 + hkv*128 + hd) -> bf16 transposed Vt[b][hkv][hd][l]
// ---------------------------------------------------------------------------
__global__ void k_v_transpose(const float* __restrict__ kvproj,
                              ushort_t* __restrict__ Vt, long total) {
  long idx = (long)blockIdx.x * blockDim.x + threadIdx.x;
  if (idx >= total) return;
  const int hd  = (int)(idx & 127);
  long rem      = idx >> 7;
  const int l   = (int)(rem % L_); rem /= L_;
  const int hkv = (int)(rem % HKV_);
  const int b   = (int)(rem / HKV_);
  const float v = kvproj[((size_t)b * L_ + l) * (2 * HKV_ * HD_) +
                         (HKV_ * HD_) + hkv * HD_ + hd];
  Vt[(((size_t)b * HKV_ + hkv) * HD_ + hd) * L_ + l] = f2bf(v);
}

// ---------------------------------------------------------------------------
// 5) Flash attention (causal, GQA). One wave = 16 query rows, full HD=128.
//    4 waves / block (64 queries). K/V read straight from global (L2-resident).
//    Q pre-scaled by 1/sqrt(HD); K laid out [b][hkv][l][hd]; V as [b][hkv][hd][l].
// ---------------------------------------------------------------------------
__global__ __launch_bounds__(128, 1)
void k_flash_attn(const ushort_t* __restrict__ Q,
                  const ushort_t* __restrict__ Kv,
                  const ushort_t* __restrict__ Vt,
                  ushort_t* __restrict__ Og) {
  const int lane  = threadIdx.x & 31;
  const int nlane = lane & 15;
  const int khalf = lane >> 4;   // "half" for both A/B frags and C layout
  const int w     = threadIdx.x >> 5;

  constexpr int nQB = L_ / 64;   // query blocks per (b,h)
  const int qb = blockIdx.x % nQB;
  const int bh = blockIdx.x / nQB;
  const int h  = bh % H_;
  const int b  = bh / H_;
  const int hkv = h >> 2;        // GROUPS = 4
  const int q0 = qb * 64 + w * 16;

  __shared__ __align__(16) ushort_t pst[4][16][64];  // per-wave P staging

  // Load Q A-fragments (16 rows x 128)
  const ushort_t* qrow = Q + (((size_t)b * H_ + h) * L_ + q0 + nlane) * HD_;
  v16bf qf[4];
#pragma unroll
  for (int kf = 0; kf < 4; ++kf) qf[kf] = load_frag(qrow, kf * 32 + khalf * 8);

  const ushort_t* Kbase = Kv + ((size_t)b * HKV_ + hkv) * L_ * HD_;
  const ushort_t* Vbase = Vt + ((size_t)b * HKV_ + hkv) * (size_t)HD_ * L_;

  v8f o[8];
#pragma unroll
  for (int j = 0; j < 8; ++j) o[j] = zero8();
  float mrun[8], lrun[8];
#pragma unroll
  for (int i = 0; i < 8; ++i) { mrun[i] = -1e30f; lrun[i] = 0.0f; }

  const int kbmax = (q0 + 15) >> 6;
  for (int kb = 0; kb <= kbmax; ++kb) {
    // ---- S = Q @ K^T for 64 keys -------------------------------------
    v8f s[4];
#pragma unroll
    for (int t = 0; t < 4; ++t) s[t] = zero8();
#pragma unroll
    for (int t = 0; t < 4; ++t) {
      const ushort_t* krow = Kbase + (size_t)(kb * 64 + t * 16 + nlane) * HD_;
#pragma unroll
      for (int kf = 0; kf < 4; ++kf)
        s[t] = wmma_bf16(qf[kf], load_frag(krow, kf * 32 + khalf * 8), s[t]);
    }

    // ---- causal mask on diagonal block -------------------------------
    if (kb == kbmax) {
#pragma unroll
      for (int t = 0; t < 4; ++t) {
        const int key = kb * 64 + t * 16 + nlane;
#pragma unroll
        for (int i = 0; i < 8; ++i)
          if (key > q0 + khalf * 8 + i) s[t][i] = -1e30f;
      }
    }

    // ---- online softmax ----------------------------------------------
    float mnew[8], alpha[8];
#pragma unroll
    for (int i = 0; i < 8; ++i) {
      float x = fmaxf(fmaxf(s[0][i], s[1][i]), fmaxf(s[2][i], s[3][i]));
      x = fmaxf(x, __shfl_xor(x, 1, 32));
      x = fmaxf(x, __shfl_xor(x, 2, 32));
      x = fmaxf(x, __shfl_xor(x, 4, 32));
      x = fmaxf(x, __shfl_xor(x, 8, 32));
      mnew[i]  = fmaxf(mrun[i], x);
      alpha[i] = __expf(mrun[i] - mnew[i]);
      mrun[i]  = mnew[i];
    }
#pragma unroll
    for (int i = 0; i < 8; ++i) {
      float sum = 0.0f;
#pragma unroll
      for (int t = 0; t < 4; ++t) {
        const float p = __expf(s[t][i] - mnew[i]);
        s[t][i] = p;
        sum += p;
      }
      sum += __shfl_xor(sum, 1, 32);
      sum += __shfl_xor(sum, 2, 32);
      sum += __shfl_xor(sum, 4, 32);
      sum += __shfl_xor(sum, 8, 32);
      lrun[i] = lrun[i] * alpha[i] + sum;
    }
#pragma unroll
    for (int j = 0; j < 8; ++j)
#pragma unroll
      for (int i = 0; i < 8; ++i) o[j][i] *= alpha[i];

    // ---- transpose P via LDS: C-layout -> A-fragments ----------------
#pragma unroll
    for (int t = 0; t < 4; ++t)
#pragma unroll
      for (int i = 0; i < 8; ++i)
        pst[w][khalf * 8 + i][t * 16 + nlane] = f2bf(s[t][i]);

    v16bf pf[2];
#pragma unroll
    for (int kf2 = 0; kf2 < 2; ++kf2)
      pf[kf2] = load_frag(&pst[w][nlane][0], kf2 * 32 + khalf * 8);

    // ---- O += P @ V  (V^T rows are hd-major, key-contiguous) ---------
#pragma unroll
    for (int t2 = 0; t2 < 8; ++t2) {
      const ushort_t* vrow = Vbase + (size_t)(t2 * 16 + nlane) * L_ + kb * 64;
#pragma unroll
      for (int kf2 = 0; kf2 < 2; ++kf2)
        o[t2] = wmma_bf16(pf[kf2], load_frag(vrow, kf2 * 32 + khalf * 8), o[t2]);
    }
  }

  // ---- normalize and store bf16 attn output as (B*L, H*HD) ----------
  float inv[8];
#pragma unroll
  for (int i = 0; i < 8; ++i) inv[i] = 1.0f / lrun[i];
#pragma unroll
  for (int t2 = 0; t2 < 8; ++t2)
#pragma unroll
    for (int i = 0; i < 8; ++i) {
      const size_t row = (size_t)b * L_ + q0 + khalf * 8 + i;
      Og[row * (size_t)D_ + h * HD_ + t2 * 16 + nlane] = f2bf(o[t2][i] * inv[i]);
    }
}

// ---------------------------------------------------------------------------
// Host launcher
// ---------------------------------------------------------------------------
extern "C" void kernel_launch(void* const* d_in, const int* in_sizes, int n_in,
                              void* d_out, int out_size, void* d_ws, size_t ws_size,
                              hipStream_t stream) {
  const float* x    = (const float*)d_in[0];
  const float* cosT = (const float*)d_in[1];
  const float* sinT = (const float*)d_in[2];
  const float* Wq   = (const float*)d_in[3];
  const float* bq   = (const float*)d_in[4];
  const float* Wkv  = (const float*)d_in[5];
  const float* bkv  = (const float*)d_in[6];
  const float* Wo   = (const float*)d_in[7];
  const float* bo   = (const float*)d_in[8];
  float* out = (float*)d_out;

  // Workspace carve-up
  char* ws = (char*)d_ws;
  size_t off = 0;
  auto take = [&](size_t bytes) { char* p = ws + off; off += (bytes + 255) & ~(size_t)255; return p; };
  const long nX   = M_ROWS * D_;                 // 8,388,608
  const long nWq  = (long)D_ * D_;               // 4,194,304
  const long nWkv = (long)(2 * HKV_ * HD_) * D_; // 2,097,152
  const long nWo  = (long)D_ * D_;
  ushort_t* xbf    = (ushort_t*)take(nX * 2);
  ushort_t* wqbf   = (ushort_t*)take(nWq * 2);
  ushort_t* wkvbf  = (ushort_t*)take(nWkv * 2);
  ushort_t* wobf   = (ushort_t*)take(nWo * 2);
  float*    qproj  = (float*)take(M_ROWS * (long)D_ * 4);
  float*    kvproj = (float*)take(M_ROWS * (long)(2 * HKV_ * HD_) * 4);
  ushort_t* Qbf    = (ushort_t*)take((long)B_ * H_ * L_ * HD_ * 2);
  ushort_t* Kbf    = (ushort_t*)take((long)B_ * HKV_ * L_ * HD_ * 2);
  ushort_t* Vtb    = (ushort_t*)take((long)B_ * HKV_ * HD_ * L_ * 2);
  ushort_t* attnO  = (ushort_t*)take(M_ROWS * (long)D_ * 2);
  (void)off; (void)ws_size; (void)n_in; (void)in_sizes; (void)out_size;

  // 1) convert inputs to bf16
  k_f32_to_bf16<<<2048, 256, 0, stream>>>(x,   xbf,   nX);
  k_f32_to_bf16<<<2048, 256, 0, stream>>>(Wq,  wqbf,  nWq);
  k_f32_to_bf16<<<1024, 256, 0, stream>>>(Wkv, wkvbf, nWkv);
  k_f32_to_bf16<<<2048, 256, 0, stream>>>(Wo,  wobf,  nWo);

  // 2) Q projection: qproj = x @ Wq^T + bq   (M=4096, N=2048, K=2048)
  {
    int M = (int)M_ROWS, N = D_, K = D_;
    int waves = (M >> 5) * (N >> 6);
    k_gemm_bf16<<<waves / 4, 128, 0, stream>>>(xbf, wqbf, bq, qproj, M, N, K);
  }
  // 3) KV projection: kvproj = x @ Wkv^T + bkv  (N=1024)
  {
    int M = (int)M_ROWS, N = 2 * HKV_ * HD_, K = D_;
    int waves = (M >> 5) * (N >> 6);
    k_gemm_bf16<<<waves / 4, 128, 0, stream>>>(xbf, wkvbf, bkv, kvproj, M, N, K);
  }

  // 4) RoPE + pack Q (scale folded) and K; V transpose
  {
    long totQ = (long)B_ * L_ * H_ * 64;
    k_rope_pack<<<(int)((totQ + 255) / 256), 256, 0, stream>>>(
        qproj, D_, cosT, sinT, Qbf, H_, QK_SCALE, totQ);
    long totK = (long)B_ * L_ * HKV_ * 64;
    k_rope_pack<<<(int)((totK + 255) / 256), 256, 0, stream>>>(
        kvproj, 2 * HKV_ * HD_, cosT, sinT, Kbf, HKV_, 1.0f, totK);
    long totV = (long)B_ * HKV_ * L_ * HD_;
    k_v_transpose<<<(int)((totV + 255) / 256), 256, 0, stream>>>(kvproj, Vtb, totV);
  }

  // 5) Flash attention: 1 block = 64 queries of one (b,h)
  {
    int blocks = B_ * H_ * (L_ / 64);  // 1024
    k_flash_attn<<<blocks, 128, 0, stream>>>(Qbf, Kbf, Vtb, attnO);
  }

  // 6) Output projection: out = attnO @ Wo^T + bo  (fp32)
  {
    int M = (int)M_ROWS, N = D_, K = D_;
    int waves = (M >> 5) * (N >> 6);
    k_gemm_bf16<<<waves / 4, 128, 0, stream>>>(attnO, wobf, bo, out, M, N, K);
  }
}
